// VQModel_LLaMA_489626272169
// MI455X (gfx1250) — matrix operand order repaired and checked by reading, not compiled, not run on hardware
//
#include <hip/hip_runtime.h>
#include <hip/hip_bf16.h>
#include <stdint.h>

// ---------------------------------------------------------------------------
// VQ quantizer for MI455X (gfx1250, wave32, WMMA).
//   z:[16,512,32,32] f32, codebook:[16384,512] f32, proj_w:[512,512] f32,
//   proj_b:[512] f32.
//   out = [ z_q (16*512*1024 f32) | loss (1 f32) | labels (16384 f32) ]
// Dominant work: 16384x16384x512 distance GEMM -> V_WMMA_F32_16X16X32_BF16.
// v3b: K3 = 32 tokens/wave (two A slabs in VGPRs, 2 WMMAs per LDS B-read)
//      + double-buffered LDS (2 x 32KB ping-pong) so the next emb chunk's
//      global loads overlap WMMA compute; one barrier per chunk.
//      (fix: no LDS-pointer aggregate -> select buffers via ternary)
// ---------------------------------------------------------------------------

typedef __bf16 bf16_t;
typedef __attribute__((ext_vector_type(16))) __bf16 v16bf;
typedef __attribute__((ext_vector_type(8)))  float  v8f;

#define T_TOK 16384          // B*H*W tokens
#define CDIM  512            // embed dim (= K of both GEMMs)
#define NCB   16384          // codebook entries
#define BETA  0.33f

static constexpr size_t ZQ_ELEMS = (size_t)16 * 512 * 1024;   // 8388608
static constexpr float  LOSS_SCALE = (1.0f + BETA) / (float)ZQ_ELEMS;

// ---- workspace layout (bytes) ----
static constexpr size_t OFF_ZF   = 0;                                   // zf  bf16 [T,512]
static constexpr size_t OFF_CBB  = OFF_ZF   + (size_t)T_TOK * CDIM * 2; // cb  bf16 [N,512]
static constexpr size_t OFF_WB   = OFF_CBB  + (size_t)NCB   * CDIM * 2; // W   bf16 [512,512]
static constexpr size_t OFF_EMBF = OFF_WB   + (size_t)CDIM  * CDIM * 2; // emb f32  [N,512]
static constexpr size_t OFF_EMBB = OFF_EMBF + (size_t)NCB   * CDIM * 4; // emb bf16 [N,512]
static constexpr size_t OFF_NORM = OFF_EMBB + (size_t)NCB   * CDIM * 2; // ||e||^2  [N]
static constexpr size_t OFF_IDX  = OFF_NORM + (size_t)NCB * 4;          // argmin   [T]

// ---------------------------------------------------------------------------
// K0: f32 -> bf16 bulk convert (n4 = n/4 float4 groups)
// ---------------------------------------------------------------------------
__global__ void k_cvt_bf16(const float* __restrict__ src,
                           bf16_t* __restrict__ dst, int n4) {
  int i = blockIdx.x * blockDim.x + threadIdx.x;
  if (i < n4) {
    float4 f = ((const float4*)src)[i];
    union { bf16_t h[4]; uint2 u; } p;
    p.h[0] = (bf16_t)f.x; p.h[1] = (bf16_t)f.y;
    p.h[2] = (bf16_t)f.z; p.h[3] = (bf16_t)f.w;
    ((uint2*)dst)[i] = p.u;
  }
}

// ---------------------------------------------------------------------------
// K1: z [b][c][hw] f32  ->  zf [t=b*1024+hw][c] bf16   (32x32 LDS transpose)
// ---------------------------------------------------------------------------
__global__ void k_pack_z(const float* __restrict__ z, bf16_t* __restrict__ zf) {
  __shared__ float tile[32][33];
  const int b  = blockIdx.z;
  const int c0 = blockIdx.y * 32;
  const int h0 = blockIdx.x * 32;
  const int tx = threadIdx.x, ty = threadIdx.y;       // 32 x 8
  const float* zp = z + (size_t)b * 512 * 1024;
#pragma unroll
  for (int i = 0; i < 4; ++i) {
    int c = c0 + ty + i * 8;
    tile[ty + i * 8][tx] = zp[(size_t)c * 1024 + h0 + tx];   // coalesced on hw
  }
  __syncthreads();
  bf16_t* out = zf + (size_t)b * 1024 * 512;
#pragma unroll
  for (int i = 0; i < 4; ++i) {
    int t = h0 + ty + i * 8;
    out[(size_t)t * 512 + c0 + tx] = (bf16_t)tile[tx][ty + i * 8]; // coalesced on c
  }
}

// ---------------------------------------------------------------------------
// WMMA operand loaders (contiguous k order, same for A and B)
// A 16x32: lane<16 row M=lane, elems K=k..k+15 contiguous; lanes 16-31 K+=16
// ---------------------------------------------------------------------------
__device__ __forceinline__ void load_a_regs(const bf16_t* __restrict__ base,
                                            int row, int hlf, v16bf* a) {
  const bf16_t* rp = base + (size_t)row * CDIM;
#pragma unroll
  for (int s = 0; s < 16; ++s) {
    const int kb = 32 * s;
    union { uint4 u[2]; v16bf v; } t;
    t.u[0] = *(const uint4*)(rp + kb + hlf * 16);
    t.u[1] = *(const uint4*)(rp + kb + 8 + hlf * 16);
    a[s] = t.v;
  }
}

// B 32x16: lane<16 col N=lane holds K=kb..kb+15 contiguous; lanes 16-31 K+=16
__device__ __forceinline__ v16bf load_b_lds(const bf16_t* __restrict__ sB,
                                            int row_local, int kb, int hlf) {
  const bf16_t* p = sB + (size_t)row_local * CDIM + kb + hlf * 16;
  union { uint4 u[2]; v16bf v; } t;
  t.u[0] = *(const uint4*)(p);
  t.u[1] = *(const uint4*)(p + 8);
  return t.v;
}

// ---------------------------------------------------------------------------
// K2: emb = codebook @ W^T + b ; also bf16 copy and row norms ||emb_n||^2.
// 128 cb-rows / block (8 waves x 16 rows, A in regs), W streamed via 64KB LDS.
// ---------------------------------------------------------------------------
__global__ __launch_bounds__(256)
void k_project(const bf16_t* __restrict__ cbb, const bf16_t* __restrict__ wb,
               const float* __restrict__ bias, float* __restrict__ embf,
               bf16_t* __restrict__ embb, float* __restrict__ norm) {
  extern __shared__ char smem[];
  bf16_t* sB = (bf16_t*)smem;                       // 64 rows x 512 bf16 = 64KB
  const int tid  = threadIdx.x;
  const int wave = tid >> 5, lane = tid & 31;
  const int l15  = lane & 15, hlf = lane >> 4;
  const int m_base = blockIdx.x * 128 + wave * 16;

  v16bf a[16];
  load_a_regs(cbb, m_base + l15, hlf, a);

  float nrm[8];
#pragma unroll
  for (int v = 0; v < 8; ++v) nrm[v] = 0.0f;

  for (int ch = 0; ch < 8; ++ch) {                  // 8 chunks of 64 e-rows
    __syncthreads();
    const uint4* src = (const uint4*)(wb + (size_t)ch * 64 * CDIM);
    uint4* dst = (uint4*)sB;
#pragma unroll
    for (int j = 0; j < 16; ++j) dst[tid + j * 256] = src[tid + j * 256];
    __syncthreads();

#pragma unroll
    for (int nt = 0; nt < 4; ++nt) {
      v8f acc = {};
#pragma unroll
      for (int s = 0; s < 16; ++s) {
        v16bf bm = load_b_lds(sB, nt * 16 + l15, 32 * s, hlf);
        acc = __builtin_amdgcn_wmma_f32_16x16x32_bf16(
            false, a[s], false, bm, (short)0, acc, false, false);
      }
      const int e  = ch * 64 + nt * 16 + l15;
      const float be = bias[e];
#pragma unroll
      for (int v = 0; v < 8; ++v) {
        const float val = acc[v] + be;
        const int m = m_base + v + 8 * hlf;
        embf[(size_t)m * CDIM + e] = val;
        embb[(size_t)m * CDIM + e] = (bf16_t)val;
        nrm[v] += val * val;
      }
    }
  }
  // sum-reduce norms across the 16 lanes of each half
#pragma unroll
  for (int msk = 1; msk < 16; msk <<= 1)
#pragma unroll
    for (int v = 0; v < 8; ++v) nrm[v] += __shfl_xor(nrm[v], msk, 32);
  if (l15 == 0) {
#pragma unroll
    for (int v = 0; v < 8; ++v) norm[m_base + v + 8 * hlf] = nrm[v];
  }
}

// ---------------------------------------------------------------------------
// K3: fused distance GEMM + argmin. 256 tokens / block; each wave owns 32
// tokens = TWO 16x512 A slabs in VGPRs, so each LDS B-read feeds 2 WMMAs.
// emb streamed through ping-pong 2 x 32KB LDS buffers: the next chunk is
// staged into VGPRs while WMMAs run on the current buffer -> one barrier
// per chunk, no workgroup-wide load stall.
// score(t,n) = ||e_n||^2 - 2 * <z_t, e_n>   (||z||^2 constant per t: dropped)
// ---------------------------------------------------------------------------
__global__ __launch_bounds__(256)
void k_argmin(const bf16_t* __restrict__ zf, const bf16_t* __restrict__ embb,
              const float* __restrict__ norm, uint32_t* __restrict__ idx_out,
              float* __restrict__ labels_out) {
  extern __shared__ char smem[];
  bf16_t* s0 = (bf16_t*)smem;                       // buffer 0: 32KB
  bf16_t* s1 = (bf16_t*)smem + 32 * CDIM;           // buffer 1: 32KB
  const int tid  = threadIdx.x;
  const int wave = tid >> 5, lane = tid & 31;
  const int l15  = lane & 15, hlf = lane >> 4;
  const int m_base = blockIdx.x * 256 + wave * 32;  // 32 tokens per wave

  v16bf a0[16], a1[16];
  load_a_regs(zf, m_base + l15,      hlf, a0);
  load_a_regs(zf, m_base + 16 + l15, hlf, a1);

  float    minv0[8], minv1[8];
  uint32_t mini0[8], mini1[8];
#pragma unroll
  for (int v = 0; v < 8; ++v) {
    minv0[v] = 3.4e38f; mini0[v] = 0u;
    minv1[v] = 3.4e38f; mini1[v] = 0u;
  }

  const int NCHUNK = NCB / 32;                      // 512 chunks of 32 codes
  // prologue: stage + store chunk 0 into buffer 0
  {
    const uint4* src = (const uint4*)embb;
    uint4* dst = (uint4*)s0;
#pragma unroll
    for (int j = 0; j < 8; ++j) dst[tid + j * 256] = src[tid + j * 256];
  }
  __syncthreads();

  for (int ch = 0; ch < NCHUNK; ++ch) {
    const bf16_t* cur = (ch & 1) ? s1 : s0;
    bf16_t*       nxt = (ch & 1) ? s0 : s1;

    // stage next chunk into VGPRs (loads overlap the WMMAs below)
    uint4 stage[8];
    if (ch + 1 < NCHUNK) {
      const uint4* src = (const uint4*)(embb + (size_t)(ch + 1) * 32 * CDIM);
#pragma unroll
      for (int j = 0; j < 8; ++j) stage[j] = src[tid + j * 256];
    }

#pragma unroll
    for (int nt = 0; nt < 2; ++nt) {
      v8f acc0 = {}, acc1 = {};
#pragma unroll
      for (int s = 0; s < 16; ++s) {
        v16bf bm = load_b_lds(cur, nt * 16 + l15, 32 * s, hlf);
        acc0 = __builtin_amdgcn_wmma_f32_16x16x32_bf16(
            false, a0[s], false, bm, (short)0, acc0, false, false);
        acc1 = __builtin_amdgcn_wmma_f32_16x16x32_bf16(
            false, a1[s], false, bm, (short)0, acc1, false, false);
      }
      const uint32_t n  = (uint32_t)(ch * 32 + nt * 16 + l15);
      const float    nn = norm[n];
#pragma unroll
      for (int v = 0; v < 8; ++v) {
        const float d0 = nn - 2.0f * acc0[v];
        if (d0 < minv0[v]) { minv0[v] = d0; mini0[v] = n; }
        const float d1 = nn - 2.0f * acc1[v];
        if (d1 < minv1[v]) { minv1[v] = d1; mini1[v] = n; }
      }
    }

    // commit staged chunk to the other buffer; single barrier per chunk:
    // after it, everyone is done reading `cur`, so chunk ch+2 may overwrite it.
    if (ch + 1 < NCHUNK) {
      uint4* dst = (uint4*)nxt;
#pragma unroll
      for (int j = 0; j < 8; ++j) dst[tid + j * 256] = stage[j];
    }
    __syncthreads();
  }

  // min-reduce (value, then lower index on ties) across 16 lanes per half
#pragma unroll
  for (int msk = 1; msk < 16; msk <<= 1) {
#pragma unroll
    for (int v = 0; v < 8; ++v) {
      float    ov = __shfl_xor(minv0[v], msk, 32);
      uint32_t oi = (uint32_t)__shfl_xor((int)mini0[v], msk, 32);
      if (ov < minv0[v] || (ov == minv0[v] && oi < mini0[v])) {
        minv0[v] = ov; mini0[v] = oi;
      }
      ov = __shfl_xor(minv1[v], msk, 32);
      oi = (uint32_t)__shfl_xor((int)mini1[v], msk, 32);
      if (ov < minv1[v] || (ov == minv1[v] && oi < mini1[v])) {
        minv1[v] = ov; mini1[v] = oi;
      }
    }
  }
  if (l15 == 0) {
#pragma unroll
    for (int v = 0; v < 8; ++v) {
      const int m0 = m_base + v + 8 * hlf;           // token ids
      const int m1 = m_base + 16 + v + 8 * hlf;
      idx_out[m0]    = mini0[v];
      labels_out[m0] = (float)mini0[v];
      idx_out[m1]    = mini1[v];
      labels_out[m1] = (float)mini1[v];
    }
  }
}

// ---------------------------------------------------------------------------
// K4: z_q[b][c][hw] = emb[idx[t]][c] (straight-through value), plus
// loss += (1+beta)/M * (z_q - z)^2. LDS-tiled so both sides stay coalesced.
// ---------------------------------------------------------------------------
__global__ void k_gather_loss(const float* __restrict__ z,
                              const float* __restrict__ embf,
                              const uint32_t* __restrict__ idx,
                              float* __restrict__ zq, float* __restrict__ loss) {
  __shared__ float tile[32][33];
  __shared__ float red[8];
  const int b  = blockIdx.z;
  const int c0 = blockIdx.y * 32;
  const int h0 = blockIdx.x * 32;
  const int tx = threadIdx.x, ty = threadIdx.y;      // 32 x 8
  const int t0 = b * 1024 + h0;

#pragma unroll
  for (int i = 0; i < 4; ++i) {                      // gather rows, coalesced c
    const int hwl = ty + i * 8;
    const uint32_t id = idx[t0 + hwl];
    tile[hwl][tx] = embf[(size_t)id * CDIM + c0 + tx];
  }
  __syncthreads();

  float acc = 0.0f;
  const float* zp = z  + (size_t)b * 512 * 1024;
  float*       qp = zq + (size_t)b * 512 * 1024;
#pragma unroll
  for (int i = 0; i < 4; ++i) {                      // write, coalesced hw
    const int cl = ty + i * 8;
    const float q  = tile[tx][cl];
    const float zv = zp[(size_t)(c0 + cl) * 1024 + h0 + tx];
    const float df = q - zv;
    acc += df * df;
    qp[(size_t)(c0 + cl) * 1024 + h0 + tx] = q;
  }
  // block reduction -> one atomic per block
#pragma unroll
  for (int msk = 16; msk >= 1; msk >>= 1) acc += __shfl_xor(acc, msk, 32);
  const int lin = ty * 32 + tx;
  if ((lin & 31) == 0) red[lin >> 5] = acc;
  __syncthreads();
  if (lin == 0) {
    float s = 0.0f;
#pragma unroll
    for (int i = 0; i < 8; ++i) s += red[i];
    atomicAdd(loss, s * LOSS_SCALE);
  }
}

// ---------------------------------------------------------------------------
extern "C" void kernel_launch(void* const* d_in, const int* in_sizes, int n_in,
                              void* d_out, int out_size, void* d_ws, size_t ws_size,
                              hipStream_t stream) {
  const float* z  = (const float*)d_in[0];
  const float* cb = (const float*)d_in[1];
  const float* pw = (const float*)d_in[2];
  const float* pb = (const float*)d_in[3];

  char* ws = (char*)d_ws;
  bf16_t*   zf    = (bf16_t*)(ws + OFF_ZF);
  bf16_t*   cbb   = (bf16_t*)(ws + OFF_CBB);
  bf16_t*   wb    = (bf16_t*)(ws + OFF_WB);
  float*    embf  = (float*) (ws + OFF_EMBF);
  bf16_t*   embb  = (bf16_t*)(ws + OFF_EMBB);
  float*    normv = (float*) (ws + OFF_NORM);
  uint32_t* idxv  = (uint32_t*)(ws + OFF_IDX);

  float* out    = (float*)d_out;
  float* zq     = out;
  float* loss   = out + ZQ_ELEMS;
  float* labels = out + ZQ_ELEMS + 1;

  (void)hipMemsetAsync(loss, 0, sizeof(float), stream);

  // bf16 packs
  {
    const int n4 = (NCB * CDIM) / 4;
    k_cvt_bf16<<<(n4 + 255) / 256, 256, 0, stream>>>(cb, cbb, n4);
    const int w4 = (CDIM * CDIM) / 4;
    k_cvt_bf16<<<(w4 + 255) / 256, 256, 0, stream>>>(pw, wb, w4);
  }
  // z transpose/pack: [b][c][hw] -> [t][c] bf16
  k_pack_z<<<dim3(32, 16, 16), dim3(32, 8), 0, stream>>>(z, zf);

  // projected codebook + norms (WMMA bf16)
  k_project<<<NCB / 128, 256, 64 * CDIM * sizeof(bf16_t), stream>>>(
      cbb, wb, pb, embf, embb, normv);

  // fused distance GEMM + argmin (WMMA bf16, dominant kernel)
  k_argmin<<<T_TOK / 256, 256, 64 * CDIM * sizeof(bf16_t), stream>>>(
      zf, embb, normv, idxv, labels);

  // gather + straight-through output + loss
  k_gather_loss<<<dim3(32, 16, 16), dim3(32, 8), 0, stream>>>(
      z, embf, idxv, zq, loss);
}
